// KNFP_GCN_2layer_76922864271370
// MI455X (gfx1250) — matrix operand order repaired and checked by reading, not compiled, not run on hardware
//
#include <hip/hip_runtime.h>
#include <hip/hip_bf16.h>
#include <math.h>

// ---------------- problem constants ----------------
static constexpr int BBI  = 512;        // graphs
static constexpr int NPGI = 200;        // nodes per graph
static constexpr int NNI  = BBI * NPGI; // 102400 nodes
static constexpr int EEI  = NNI * 8;    // 819200 edges
static constexpr int KP1  = 160;        // top-k 1
static constexpr int KP2  = 128;        // top-k 2
static constexpr int HHI  = 128;        // hidden
static constexpr int GHI  = 120;        // GRU hidden
static constexpr int G3I  = 360;        // 3*GH
static constexpr int GPI  = 384;        // padded 3*GH (mult of 64)
static constexpr int TTI  = 101;
static constexpr int TCI  = 99;         // conv valid length
static constexpr int TPI  = 19;         // pooled length
static constexpr int CKI  = 252;        // 84*3 conv K
static constexpr int CKPI = 256;        // padded conv K
static constexpr int CMI  = BBI * TCI;  // 50688 conv GEMM rows
static constexpr int XMI  = TPI * BBI;  // 9728 GRU GEMM rows
static constexpr int D1KI  = 4816;      // dense-1 K (2H + 2*GH*TP)
static constexpr int D1KPI = 4832;      // padded (mult of 32)
static constexpr int D1NI  = 102;
static constexpr int D1NPI = 128;       // padded (mult of 64)

// ---------------- types ----------------
typedef __bf16 bf16_t;
typedef bf16_t v16bf __attribute__((ext_vector_type(16)));
typedef bf16_t v8bf  __attribute__((ext_vector_type(8)));
typedef float  v8f   __attribute__((ext_vector_type(8)));

__device__ __forceinline__ unsigned short f2bf(float f) {
  unsigned int u = __float_as_uint(f);
  unsigned int r = (u + 0x7FFFu + ((u >> 16) & 1u)) >> 16;   // RNE
  return (unsigned short)r;
}
__device__ __forceinline__ float sigm(float x) { return 1.0f / (1.0f + __expf(-x)); }

// =====================================================================
// WMMA bf16 GEMM: C[M,Np] = A[M,Kp] * BT[Np,Kp]^T (+bias, opt relu)
// one wave -> 16(M) x 64(N) tile, K-step 32, fp32 accumulate
// speculative global_prefetch one K-tile ahead overlaps HBM/L2 latency
// with the current step's WMMAs (prefetch does not touch LOADcnt).
// =====================================================================
__global__ void k_gemm_bf16(const unsigned short* __restrict__ Au,
                            const unsigned short* __restrict__ BTu,
                            float* __restrict__ C,
                            const float* __restrict__ bias,
                            int M, int Kp, int Np, int nbias, int relu) {
  const bf16_t* A  = (const bf16_t*)Au;
  const bf16_t* BT = (const bf16_t*)BTu;
  int wid  = (blockIdx.x * blockDim.x + threadIdx.x) >> 5;
  int lane = threadIdx.x & 31;
  int tilesN = Np >> 6;
  int total  = (M >> 4) * tilesN;
  if (wid >= total) return;
  int tm = wid / tilesN, tn = wid % tilesN;
  int m0 = tm << 4, n0 = tn << 6;
  int ncol = lane & 15;
  int kh   = lane >> 4;
  int mrow = m0 + ncol;

  const bf16_t* arow = A + (size_t)mrow * Kp + kh * 16;
  const bf16_t* brow = BT + (size_t)(n0 + ncol) * Kp + kh * 16;

  v8f c0 = {0,0,0,0,0,0,0,0};
  v8f c1 = {0,0,0,0,0,0,0,0};
  v8f c2 = {0,0,0,0,0,0,0,0};
  v8f c3 = {0,0,0,0,0,0,0,0};

  for (int k0 = 0; k0 < Kp; k0 += 32) {
    // A fragment: lane owns row m, contiguous K run of 16 (K=kh*16..+15)
    const bf16_t* ap = arow + k0;
    v8bf alo = *(const v8bf*)ap;
    v8bf ahi = *(const v8bf*)(ap + 8);
    v16bf a;
#pragma unroll
    for (int i = 0; i < 8; ++i) { a[i] = alo[i]; a[8 + i] = ahi[i]; }
    // B fragments: lane owns column n, contiguous K run of 16
    const bf16_t* bp = brow + k0;
    v16bf b0 = *(const v16bf*)bp;
    v16bf b1 = *(const v16bf*)(bp + (size_t)16 * Kp);
    v16bf b2 = *(const v16bf*)(bp + (size_t)32 * Kp);
    v16bf b3 = *(const v16bf*)(bp + (size_t)48 * Kp);
    // speculative prefetch one K-tile ahead (global_prefetch_b8)
    if (k0 + 32 < Kp) {
      __builtin_prefetch(ap + 32, 0, 1);
      __builtin_prefetch(bp + 32, 0, 1);
      __builtin_prefetch(bp + (size_t)16 * Kp + 32, 0, 1);
      __builtin_prefetch(bp + (size_t)32 * Kp + 32, 0, 1);
      __builtin_prefetch(bp + (size_t)48 * Kp + 32, 0, 1);
    }
    c0 = __builtin_amdgcn_wmma_f32_16x16x32_bf16(false, a, false, b0, (short)0, c0, false, false);
    c1 = __builtin_amdgcn_wmma_f32_16x16x32_bf16(false, a, false, b1, (short)0, c1, false, false);
    c2 = __builtin_amdgcn_wmma_f32_16x16x32_bf16(false, a, false, b2, (short)0, c2, false, false);
    c3 = __builtin_amdgcn_wmma_f32_16x16x32_bf16(false, a, false, b3, (short)0, c3, false, false);
  }

  int rbase = m0 + (kh << 3);
#define STORE_SUB(cv, s)                                                     \
  {                                                                          \
    int n = n0 + (s) * 16 + ncol;                                            \
    float bv = (bias && n < nbias) ? bias[n] : 0.0f;                         \
    _Pragma("unroll") for (int r = 0; r < 8; ++r) {                          \
      float v = cv[r] + bv;                                                  \
      if (relu) v = fmaxf(v, 0.0f);                                          \
      C[(size_t)(rbase + r) * Np + n] = v;                                   \
    }                                                                        \
  }
  STORE_SUB(c0, 0) STORE_SUB(c1, 1) STORE_SUB(c2, 2) STORE_SUB(c3, 3)
#undef STORE_SUB
}

// =====================================================================
// glue kernels
// =====================================================================
__global__ void k_msg1(const float* __restrict__ x, const float* __restrict__ w,
                       float* __restrict__ out) {
  int idx = blockIdx.x * blockDim.x + threadIdx.x;
  if (idx >= NNI * HHI) return;
  int f = idx & (HHI - 1), i = idx >> 7;
  const float* xr = x + (size_t)i * 4;
  out[idx] = xr[0]*w[0*HHI+f] + xr[1]*w[1*HHI+f] + xr[2]*w[2*HHI+f] + xr[3]*w[3*HHI+f];
}

__global__ void k_scatter(const float* __restrict__ msg, const int* __restrict__ src,
                          const int* __restrict__ dst, const int* __restrict__ mask,
                          float* __restrict__ agg) {
  long long idx = (long long)blockIdx.x * blockDim.x + threadIdx.x;
  if (idx >= (long long)EEI * HHI) return;
  int f = (int)(idx & (HHI - 1)), e = (int)(idx >> 7);
  int s = src[e], d = dst[e];
  if (mask && (!mask[s] || !mask[d])) return;
  atomicAdd(&agg[(size_t)d * HHI + f], msg[(size_t)s * HHI + f]);
}

__global__ void k_node1(const float* __restrict__ agg, const float* __restrict__ x,
                        const float* __restrict__ wroot, const float* __restrict__ b,
                        float* __restrict__ h) {
  int idx = blockIdx.x * blockDim.x + threadIdx.x;
  if (idx >= NNI * HHI) return;
  int f = idx & (HHI - 1), i = idx >> 7;
  const float* xr = x + (size_t)i * 4;
  float root = xr[0]*wroot[0*HHI+f] + xr[1]*wroot[1*HHI+f] +
               xr[2]*wroot[2*HHI+f] + xr[3]*wroot[3*HHI+f];
  h[idx] = fmaxf(agg[idx] + root + b[f], 0.0f);
}

__global__ void k_node2(const float* __restrict__ agg, float* __restrict__ hroot,
                        const float* __restrict__ b) {
  int idx = blockIdx.x * blockDim.x + threadIdx.x;
  if (idx >= NNI * HHI) return;
  int f = idx & (HHI - 1);
  hroot[idx] = fmaxf(agg[idx] + hroot[idx] + b[f], 0.0f);
}

__global__ void k_score(const float* __restrict__ h, const float* __restrict__ p,
                        float* __restrict__ s) {
  int i = blockIdx.x * blockDim.x + threadIdx.x;
  if (i >= NNI) return;
  float pp = 0.0f, d = 0.0f;
  for (int k = 0; k < HHI; ++k) { pp += p[k]*p[k]; d += h[(size_t)i*HHI+k]*p[k]; }
  s[i] = d * rsqrtf(pp);
}

__global__ void k_topk(const float* __restrict__ score, const int* __restrict__ maskIn,
                       int* __restrict__ maskOut, int K) {
  __shared__ float ss[NPGI];
  int b = blockIdx.x;
  for (int i = threadIdx.x; i < NPGI; i += blockDim.x) {
    int gi = b * NPGI + i;
    bool ok = (maskIn == nullptr) || (maskIn[gi] != 0);
    ss[i] = ok ? score[gi] : -__builtin_inff();
  }
  __syncthreads();
  for (int i = threadIdx.x; i < NPGI; i += blockDim.x) {
    float si = ss[i];
    int rank = 0;
    for (int j = 0; j < NPGI; ++j) {
      float sj = ss[j];
      rank += (sj > si) || (sj == si && j < i);
    }
    maskOut[b * NPGI + i] = (si > -__builtin_inff() && rank < K) ? 1 : 0;
  }
}

__global__ void k_scale(float* __restrict__ h, const float* __restrict__ score) {
  int idx = blockIdx.x * blockDim.x + threadIdx.x;
  if (idx >= NNI * HHI) return;
  h[idx] *= tanhf(score[idx >> 7]);
}

__global__ void k_readout(const float* __restrict__ h, const int* __restrict__ mask,
                          float* __restrict__ out, float kk) {
  int b = blockIdx.x, f = threadIdx.x;   // 128 threads
  float mx = -__builtin_inff(), sm = 0.0f;
  for (int i = 0; i < NPGI; ++i) {
    if (mask[b * NPGI + i]) {
      float v = h[((size_t)b * NPGI + i) * HHI + f];
      mx = fmaxf(mx, v); sm += v;
    }
  }
  out[b * 256 + f] = mx;
  out[b * 256 + 128 + f] = sm / kk;
}

__global__ void k_f2bf(const float* __restrict__ src, unsigned short* __restrict__ dst, int n) {
  int i = blockIdx.x * blockDim.x + threadIdx.x;
  if (i < n) dst[i] = f2bf(src[i]);
}

// pack B^T from row-major [K,N] source:  dst[n*Kp+k] = src[k*N+n]
__global__ void k_pack_bt(const float* __restrict__ src, unsigned short* __restrict__ dst,
                          int K, int Nr, int Kp, int Np) {
  int idx = blockIdx.x * blockDim.x + threadIdx.x;
  if (idx >= Np * Kp) return;
  int k = idx % Kp, n = idx / Kp;
  float v = (k < K && n < Nr) ? src[(size_t)k * Nr + n] : 0.0f;
  dst[idx] = f2bf(v);
}

// pack from already-[N,K] row-major source (cw, wif, wib), zero-padded
__global__ void k_pack_rm(const float* __restrict__ src, unsigned short* __restrict__ dst,
                          int K, int Nr, int Kp, int Np) {
  int idx = blockIdx.x * blockDim.x + threadIdx.x;
  if (idx >= Np * Kp) return;
  int k = idx % Kp, n = idx / Kp;
  float v = (k < K && n < Nr) ? src[(size_t)n * K + k] : 0.0f;
  dst[idx] = f2bf(v);
}

__global__ void k_im2col(const float* __restrict__ tgt, unsigned short* __restrict__ dst) {
  int idx = blockIdx.x * blockDim.x + threadIdx.x;
  if (idx >= CMI * CKPI) return;
  int col = idx % CKPI, row = idx / CKPI;
  int t = row % TCI, b = row / TCI;
  float v = 0.0f;
  if (col < CKI) {
    int c = col / 3, kk = col % 3;
    v = tgt[((size_t)b * 84 + c) * TTI + t + kk];
  }
  dst[idx] = f2bf(v);
}

__global__ void k_pool(const float* __restrict__ xl, unsigned short* __restrict__ X) {
  int idx = blockIdx.x * blockDim.x + threadIdx.x;
  if (idx >= XMI * HHI) return;
  int f = idx & 127, r = idx >> 7;
  int b = r % BBI, t = r / BBI;
  float s = 0.0f;
#pragma unroll
  for (int q = 0; q < 5; ++q) s += xl[((size_t)b * TCI + t * 5 + q) * HHI + f];
  X[idx] = f2bf(s * 0.2f);
}

__global__ void k_gru_step(const float* __restrict__ gi, int tg,
                           const float* __restrict__ hprev, float* __restrict__ hout,
                           const float* __restrict__ whh, const float* __restrict__ bhh) {
  int idx = blockIdx.x * blockDim.x + threadIdx.x;
  if (idx >= BBI * GHI) return;
  int j = idx % GHI, b = idx / GHI;
  const float* hp = hprev + (size_t)b * GHI;
  float ghr = bhh[j], ghz = bhh[GHI + j], ghn = bhh[2 * GHI + j];
  const float* wr = whh + (size_t)j * GHI;
  const float* wz = whh + (size_t)(GHI + j) * GHI;
  const float* wn = whh + (size_t)(2 * GHI + j) * GHI;
  for (int k = 0; k < GHI; ++k) {
    float hk = hp[k];
    ghr += hk * wr[k]; ghz += hk * wz[k]; ghn += hk * wn[k];
  }
  size_t base = ((size_t)tg * BBI + b) * GPI;
  float r = sigm(gi[base + j] + ghr);
  float z = sigm(gi[base + GHI + j] + ghz);
  float n = tanhf(gi[base + 2 * GHI + j] + r * ghn);
  hout[idx] = (1.0f - z) * n + z * hp[j];
}

__global__ void k_feat(const float* __restrict__ x1, const float* __restrict__ x2,
                       const float* __restrict__ hf, const float* __restrict__ hb,
                       unsigned short* __restrict__ feat) {
  int idx = blockIdx.x * blockDim.x + threadIdx.x;
  if (idx >= BBI * D1KPI) return;
  int col = idx % D1KPI, b = idx / D1KPI;
  float v = 0.0f;
  if (col < 256) {
    v = x1[b * 256 + col] + x2[b * 256 + col];
  } else if (col < D1KI) {
    int cc = col - 256;
    int t = cc / 240, j = cc % 240;
    v = (j < GHI) ? hf[((size_t)t * BBI + b) * GHI + j]
                  : hb[((size_t)t * BBI + b) * GHI + (j - GHI)];
  }
  feat[idx] = f2bf(v);
}

__global__ void k_head(const float* __restrict__ xc, const float* __restrict__ d3w,
                       const float* __restrict__ d3b, float* __restrict__ out) {
  int b = blockIdx.x * blockDim.x + threadIdx.x;
  if (b >= BBI) return;
  float l0 = d3b[0], l1 = d3b[1];
  for (int k = 0; k < D1NI; ++k) {
    float v = xc[(size_t)b * D1NPI + k];
    l0 += v * d3w[k * 2 + 0];
    l1 += v * d3w[k * 2 + 1];
  }
  float m = fmaxf(l0, l1);
  float lse = m + logf(expf(l0 - m) + expf(l1 - m));
  out[b * 2 + 0] = l0 - lse;
  out[b * 2 + 1] = l1 - lse;
}

// =====================================================================
// host orchestration
// =====================================================================
static inline int cdiv(long long a, long long b) { return (int)((a + b - 1) / b); }

static inline void gemm(const void* A, const void* BT, float* C, const float* bias,
                        int M, int Kp, int Np, int nbias, int relu, hipStream_t s) {
  int waves = (M / 16) * (Np / 64);
  int blocks = (waves + 7) / 8;
  k_gemm_bf16<<<blocks, 256, 0, s>>>((const unsigned short*)A, (const unsigned short*)BT,
                                     C, bias, M, Kp, Np, nbias, relu);
}

extern "C" void kernel_launch(void* const* d_in, const int* in_sizes, int n_in,
                              void* d_out, int out_size, void* d_ws, size_t ws_size,
                              hipStream_t stream) {
  const float* x        = (const float*)d_in[0];
  const int*   ei       = (const int*)d_in[1];
  const int*   src      = ei;
  const int*   dst      = ei + EEI;
  const float* target01 = (const float*)d_in[3];
  const float* w1_rel   = (const float*)d_in[4];
  const float* w1_root  = (const float*)d_in[5];
  const float* b1       = (const float*)d_in[6];
  const float* p1       = (const float*)d_in[7];
  const float* w2_rel   = (const float*)d_in[8];
  const float* w2_root  = (const float*)d_in[9];
  const float* b2       = (const float*)d_in[10];
  const float* p2       = (const float*)d_in[11];
  const float* cw       = (const float*)d_in[12];
  const float* cb       = (const float*)d_in[13];
  const float* wif      = (const float*)d_in[14];
  const float* whf      = (const float*)d_in[15];
  const float* bif      = (const float*)d_in[16];
  const float* bhf      = (const float*)d_in[17];
  const float* wib      = (const float*)d_in[18];
  const float* whb      = (const float*)d_in[19];
  const float* bib      = (const float*)d_in[20];
  const float* bhb      = (const float*)d_in[21];
  const float* d1w      = (const float*)d_in[22];
  const float* d1b      = (const float*)d_in[23];
  const float* d3w      = (const float*)d_in[24];
  const float* d3b      = (const float*)d_in[25];
  float* out = (float*)d_out;

  // ---- workspace bump allocator ----
  size_t off = 0;
  auto alloc = [&](size_t bytes) -> char* {
    char* p = (char*)d_ws + off;
    off = (off + bytes + 255) & ~(size_t)255;
    return p;
  };
  const size_t NH4 = (size_t)NNI * HHI * 4;
  const size_t NH2 = (size_t)NNI * HHI * 2;
  float*          bufA   = (float*)alloc(NH4);                         // msg1 -> hw2rel
  float*          bufB   = (float*)alloc(NH4);                         // agg1 -> agg2
  float*          bufC   = (float*)alloc(NH4);                         // h1 -> hw2root -> h2
  unsigned short* h1bf   = (unsigned short*)alloc(NH2);
  float*          score  = (float*)alloc((size_t)NNI * 4);
  int*            mask1  = (int*)alloc((size_t)NNI * 4);
  int*            mask2  = (int*)alloc((size_t)NNI * 4);
  float*          x1     = (float*)alloc((size_t)BBI * 256 * 4);
  float*          x2     = (float*)alloc((size_t)BBI * 256 * 4);
  unsigned short* w2relBT  = (unsigned short*)alloc((size_t)HHI * HHI * 2);
  unsigned short* w2rootBT = (unsigned short*)alloc((size_t)HHI * HHI * 2);
  unsigned short* cwBT   = (unsigned short*)alloc((size_t)128 * CKPI * 2);
  unsigned short* wifBT  = (unsigned short*)alloc((size_t)GPI * HHI * 2);
  unsigned short* wibBT  = (unsigned short*)alloc((size_t)GPI * HHI * 2);
  unsigned short* d1wBT  = (unsigned short*)alloc((size_t)D1NPI * D1KPI * 2);
  unsigned short* im2    = (unsigned short*)alloc((size_t)CMI * CKPI * 2);
  float*          xl     = (float*)alloc((size_t)CMI * HHI * 4);
  unsigned short* Xbf    = (unsigned short*)alloc((size_t)XMI * HHI * 2);
  float*          gif    = (float*)alloc((size_t)XMI * GPI * 4);
  float*          gib    = (float*)alloc((size_t)XMI * GPI * 4);
  float*          hf     = (float*)alloc((size_t)TPI * BBI * GHI * 4);
  float*          hb     = (float*)alloc((size_t)TPI * BBI * GHI * 4);
  float*          hzero  = (float*)alloc((size_t)BBI * GHI * 4);
  unsigned short* featbf = (unsigned short*)alloc((size_t)BBI * D1KPI * 2);
  float*          xc     = (float*)alloc((size_t)BBI * D1NPI * 4);

  const int TB = 256;
  // ---------- GraphConv layer 1 ----------
  hipMemsetAsync(bufB, 0, NH4, stream);
  k_msg1<<<cdiv((long long)NNI * HHI, TB), TB, 0, stream>>>(x, w1_rel, bufA);
  k_scatter<<<cdiv((long long)EEI * HHI, TB), TB, 0, stream>>>(bufA, src, dst, nullptr, bufB);
  k_node1<<<cdiv((long long)NNI * HHI, TB), TB, 0, stream>>>(bufB, x, w1_root, b1, bufC);
  // ---------- TopK 1 + readout 1 ----------
  k_score<<<cdiv(NNI, TB), TB, 0, stream>>>(bufC, p1, score);
  k_topk<<<BBI, TB, 0, stream>>>(score, nullptr, mask1, KP1);
  k_scale<<<cdiv((long long)NNI * HHI, TB), TB, 0, stream>>>(bufC, score);
  k_readout<<<BBI, 128, 0, stream>>>(bufC, mask1, x1, (float)KP1);
  // ---------- GraphConv layer 2 (WMMA GEMMs) ----------
  k_f2bf<<<cdiv((long long)NNI * HHI, TB), TB, 0, stream>>>(bufC, h1bf, NNI * HHI);
  k_pack_bt<<<cdiv(HHI * HHI, TB), TB, 0, stream>>>(w2_rel, w2relBT, HHI, HHI, HHI, HHI);
  k_pack_bt<<<cdiv(HHI * HHI, TB), TB, 0, stream>>>(w2_root, w2rootBT, HHI, HHI, HHI, HHI);
  hipMemsetAsync(bufB, 0, NH4, stream);
  gemm(h1bf, w2relBT,  bufA, nullptr, NNI, HHI, HHI, 0, 0, stream);   // h @ w2_rel
  gemm(h1bf, w2rootBT, bufC, nullptr, NNI, HHI, HHI, 0, 0, stream);   // h @ w2_root
  k_scatter<<<cdiv((long long)EEI * HHI, TB), TB, 0, stream>>>(bufA, src, dst, mask1, bufB);
  k_node2<<<cdiv((long long)NNI * HHI, TB), TB, 0, stream>>>(bufB, bufC, b2);
  // ---------- TopK 2 + readout 2 ----------
  k_score<<<cdiv(NNI, TB), TB, 0, stream>>>(bufC, p2, score);
  k_topk<<<BBI, TB, 0, stream>>>(score, mask1, mask2, KP2);
  k_scale<<<cdiv((long long)NNI * HHI, TB), TB, 0, stream>>>(bufC, score);
  k_readout<<<BBI, 128, 0, stream>>>(bufC, mask2, x2, (float)KP2);
  // ---------- Conv1d branch as im2col GEMM ----------
  k_im2col<<<cdiv((long long)CMI * CKPI, TB), TB, 0, stream>>>(target01, im2);
  k_pack_rm<<<cdiv(128 * CKPI, TB), TB, 0, stream>>>(cw, cwBT, CKI, 128, CKPI, 128);
  gemm(im2, cwBT, xl, cb, CMI, CKPI, 128, 128, 1, stream);            // relu(conv)
  k_pool<<<cdiv((long long)XMI * HHI, TB), TB, 0, stream>>>(xl, Xbf);
  // ---------- GRU input projections (WMMA) ----------
  k_pack_rm<<<cdiv(GPI * HHI, TB), TB, 0, stream>>>(wif, wifBT, HHI, G3I, HHI, GPI);
  k_pack_rm<<<cdiv(GPI * HHI, TB), TB, 0, stream>>>(wib, wibBT, HHI, G3I, HHI, GPI);
  gemm(Xbf, wifBT, gif, bif, XMI, HHI, GPI, G3I, 0, stream);
  gemm(Xbf, wibBT, gib, bib, XMI, HHI, GPI, G3I, 0, stream);
  // ---------- GRU recurrence (sequential, latency-bound) ----------
  hipMemsetAsync(hzero, 0, (size_t)BBI * GHI * 4, stream);
  int gblk = cdiv(BBI * GHI, TB);
  for (int t = 0; t < TPI; ++t) {
    const float* hp = (t == 0) ? hzero : hf + (size_t)(t - 1) * BBI * GHI;
    k_gru_step<<<gblk, TB, 0, stream>>>(gif, t, hp, hf + (size_t)t * BBI * GHI, whf, bhf);
  }
  for (int s = 0; s < TPI; ++s) {
    int tg = TPI - 1 - s;
    const float* hp = (s == 0) ? hzero : hb + (size_t)(tg + 1) * BBI * GHI;
    k_gru_step<<<gblk, TB, 0, stream>>>(gib, tg, hp, hb + (size_t)tg * BBI * GHI, whb, bhb);
  }
  // ---------- dense head ----------
  k_feat<<<cdiv((long long)BBI * D1KPI, TB), TB, 0, stream>>>(x1, x2, hf, hb, featbf);
  k_pack_bt<<<cdiv(D1NPI * D1KPI, TB), TB, 0, stream>>>(d1w, d1wBT, D1KI, D1NI, D1KPI, D1NPI);
  gemm(featbf, d1wBT, xc, d1b, BBI, D1KPI, D1NPI, D1NI, 1, stream);
  k_head<<<cdiv(BBI, TB), TB, 0, stream>>>(xc, d3w, d3b, out);
  (void)in_sizes; (void)n_in; (void)out_size; (void)ws_size;
}
